// MultiHeadAttention_10874857193686
// MI455X (gfx1250) — matrix operand order repaired and checked by reading, not compiled, not run on hardware
//
#include <hip/hip_runtime.h>

#define SEQ    2048
#define BATCH  2
#define DM     1024
#define NH     16
#define HD     64
#define NROWS  (BATCH * SEQ)   // 4096

typedef __attribute__((ext_vector_type(16))) __bf16 v16bf;
typedef __attribute__((ext_vector_type(8)))  float  v8f;

union FragB {
    v16bf v;
    uint4 q[2];
};

__device__ __forceinline__ unsigned short f2bf(float f) {
    union { float f; unsigned u; } x;
    x.f = f;
    unsigned r = x.u + 0x7FFFu + ((x.u >> 16) & 1u);   // RNE
    return (unsigned short)(r >> 16);
}

__device__ __forceinline__ v8f wmma_bf16(const FragB& a, const FragB& b, v8f c) {
    return __builtin_amdgcn_wmma_f32_16x16x32_bf16(
        false, a.v, false, b.v, (short)0, c, false, false);
}

// ---------------- prep kernels ----------------

__global__ void k_cvt(const float* __restrict__ src, unsigned short* __restrict__ dst, int n) {
    int i = blockIdx.x * blockDim.x + threadIdx.x;
    if (i < n) dst[i] = f2bf(src[i]);
}

// pack Wq|Wk|Wv -> bf16, N-major: wT[n*1024 + k] = W[k][n%1024]
__global__ void k_packT_qkv(const float* __restrict__ Wq, const float* __restrict__ Wk,
                            const float* __restrict__ Wv, unsigned short* __restrict__ wT) {
    int i = blockIdx.x * 256 + threadIdx.x;   // over 3072*1024
    int n = i >> 10;
    int k = i & 1023;
    const float* W = (n < 1024) ? Wq : ((n < 2048) ? Wk : Wv);
    int nl = n & 1023;
    wT[i] = f2bf(W[k * 1024 + nl]);
}

__global__ void k_packT(const float* __restrict__ W, unsigned short* __restrict__ wT) {
    int i = blockIdx.x * 256 + threadIdx.x;   // over 1024*1024
    int n = i >> 10;
    int k = i & 1023;
    wT[i] = f2bf(W[k * 1024 + n]);
}

// ---------------- fused QKV GEMM + bias + RoPE ----------------
// block 256 = 8 waves; wave tile: 32 rows x 64 cols; grid (3072/64, 4096/256)

__global__ __launch_bounds__(256) void k_gemm_qkv(
    const unsigned short* __restrict__ xbf, const unsigned short* __restrict__ wT,
    const float* __restrict__ bq, const float* __restrict__ bk, const float* __restrict__ bv,
    unsigned short* __restrict__ qbf, unsigned short* __restrict__ kbf,
    unsigned short* __restrict__ vT) {

    const int wave = threadIdx.x >> 5;
    const int lane = threadIdx.x & 31;
    const int l    = lane & 15;
    const int half = lane >> 4;
    const int row_base = blockIdx.y * 256 + wave * 32;
    const int n_base   = blockIdx.x * 64;

    const v8f zero = {0.f, 0.f, 0.f, 0.f, 0.f, 0.f, 0.f, 0.f};
    v8f c[2][4];
#pragma unroll
    for (int rt = 0; rt < 2; ++rt)
#pragma unroll
        for (int t = 0; t < 4; ++t) c[rt][t] = zero;

    for (int k = 0; k < DM; k += 32) {
        FragB a[2];
#pragma unroll
        for (int rt = 0; rt < 2; ++rt) {
            const unsigned short* ap =
                xbf + (size_t)(row_base + rt * 16 + l) * DM + k + half * 8;
            a[rt].q[0] = *(const uint4*)(ap);
            a[rt].q[1] = *(const uint4*)(ap + 16);
        }
#pragma unroll
        for (int t = 0; t < 4; ++t) {
            FragB bf;
            const unsigned short* bp =
                wT + (size_t)(n_base + t * 16 + l) * DM + k + half * 8;
            bf.q[0] = *(const uint4*)(bp);
            bf.q[1] = *(const uint4*)(bp + 16);
            c[0][t] = wmma_bf16(a[0], bf, c[0][t]);
            c[1][t] = wmma_bf16(a[1], bf, c[1][t]);
        }
    }

    const int which = n_base >> 10;        // 0=q 1=k 2=v
    const int nloc  = n_base & 1023;
    const int h     = nloc >> 6;
    const float* bias = (which == 0) ? bq : ((which == 1) ? bk : bv);
    float bc[4];
#pragma unroll
    for (int t = 0; t < 4; ++t) bc[t] = bias[nloc + t * 16 + l];

    if (which < 2) {
        unsigned short* dst = (which == 0) ? qbf : kbf;
        const float fscale = (which == 0) ? 0.125f : 1.0f;   // fold 1/sqrt(64) into q
        const float if0 = powf(10000.f, -(float)l        / 32.f);
        const float if1 = powf(10000.f, -(float)(16 + l) / 32.f);
#pragma unroll
        for (int rt = 0; rt < 2; ++rt) {
#pragma unroll
            for (int j = 0; j < 8; ++j) {
                int m = row_base + rt * 16 + j + 8 * half;
                int b = m >> 11, s = m & 2047;
                float sn0, cs0, sn1, cs1;
                sincosf((float)s * if0, &sn0, &cs0);
                sincosf((float)s * if1, &sn1, &cs1);
                float e0 = c[rt][0][j] + bc[0];
                float e1 = c[rt][1][j] + bc[1];
                float e2 = c[rt][2][j] + bc[2];
                float e3 = c[rt][3][j] + bc[3];
                float r0 = (e0 * cs0 - e2 * sn0) * fscale;
                float r1 = (e1 * cs1 - e3 * sn1) * fscale;
                float r2 = (e2 * cs0 + e0 * sn0) * fscale;
                float r3 = (e3 * cs1 + e1 * sn1) * fscale;
                size_t base = ((size_t)((b * NH + h) * SEQ + s)) * HD + l;
                dst[base +  0] = f2bf(r0);
                dst[base + 16] = f2bf(r1);
                dst[base + 32] = f2bf(r2);
                dst[base + 48] = f2bf(r3);
            }
        }
    } else {
#pragma unroll
        for (int rt = 0; rt < 2; ++rt) {
#pragma unroll
            for (int j = 0; j < 8; ++j) {
                int m = row_base + rt * 16 + j + 8 * half;
                int b = m >> 11, s = m & 2047;
#pragma unroll
                for (int t = 0; t < 4; ++t) {
                    float val = c[rt][t][j] + bc[t];
                    vT[((size_t)((b * NH + h) * HD + t * 16 + l)) * SEQ + s] = f2bf(val);
                }
            }
        }
    }
}

// ---------------- flash attention ----------------
// block 128 = 4 waves; wave: 16 query rows; grid (SEQ/64, B*H)

__global__ __launch_bounds__(128) void k_attn(
    const unsigned short* __restrict__ qbf, const unsigned short* __restrict__ kbf,
    const unsigned short* __restrict__ vT, unsigned short* __restrict__ aout) {

    __shared__ __align__(16) unsigned short pshm[4][16][40];   // 80B row stride

    const int wave = threadIdx.x >> 5;
    const int lane = threadIdx.x & 31;
    const int l    = lane & 15;
    const int half = lane >> 4;
    const int bh   = blockIdx.y;
    const int b    = bh >> 4, h = bh & 15;
    const int qrow = blockIdx.x * 64 + wave * 16;

    const unsigned short* qb = qbf + (size_t)bh * SEQ * HD;
    const unsigned short* kb = kbf + (size_t)bh * SEQ * HD;
    const unsigned short* vb = vT  + (size_t)bh * HD * SEQ;

    FragB qa[2];
#pragma unroll
    for (int ks = 0; ks < 2; ++ks) {
        const unsigned short* qp = qb + (size_t)(qrow + l) * HD + ks * 32 + half * 8;
        qa[ks].q[0] = *(const uint4*)(qp);
        qa[ks].q[1] = *(const uint4*)(qp + 16);
    }

    const v8f zero = {0.f, 0.f, 0.f, 0.f, 0.f, 0.f, 0.f, 0.f};
    v8f o[4];
    float mrun[8], lrun[8];
#pragma unroll
    for (int t = 0; t < 4; ++t) o[t] = zero;
#pragma unroll
    for (int j = 0; j < 8; ++j) { mrun[j] = -1e30f; lrun[j] = 0.f; }

    for (int kblk = 0; kblk < SEQ; kblk += 32) {
        v8f s0 = zero, s1 = zero;
#pragma unroll
        for (int ks = 0; ks < 2; ++ks) {
            FragB kf0, kf1;
            const unsigned short* kp0 = kb + (size_t)(kblk + l) * HD + ks * 32 + half * 8;
            kf0.q[0] = *(const uint4*)(kp0);
            kf0.q[1] = *(const uint4*)(kp0 + 16);
            const unsigned short* kp1 = kb + (size_t)(kblk + 16 + l) * HD + ks * 32 + half * 8;
            kf1.q[0] = *(const uint4*)(kp1);
            kf1.q[1] = *(const uint4*)(kp1 + 16);
            s0 = wmma_bf16(qa[ks], kf0, s0);
            s1 = wmma_bf16(qa[ks], kf1, s1);
        }
        // online softmax per row (row j+8*half; 16 N-lanes per half-wave)
#pragma unroll
        for (int j = 0; j < 8; ++j) {
            float t = fmaxf(s0[j], s1[j]);
            t = fmaxf(t, __shfl_xor(t, 1, 32));
            t = fmaxf(t, __shfl_xor(t, 2, 32));
            t = fmaxf(t, __shfl_xor(t, 4, 32));
            t = fmaxf(t, __shfl_xor(t, 8, 32));
            float mnew  = fmaxf(mrun[j], t);
            float alpha = expf(mrun[j] - mnew);
            float p0 = expf(s0[j] - mnew);
            float p1 = expf(s1[j] - mnew);
            float rs = p0 + p1;
            rs += __shfl_xor(rs, 1, 32);
            rs += __shfl_xor(rs, 2, 32);
            rs += __shfl_xor(rs, 4, 32);
            rs += __shfl_xor(rs, 8, 32);
            lrun[j] = lrun[j] * alpha + rs;
            mrun[j] = mnew;
            o[0][j] *= alpha; o[1][j] *= alpha; o[2][j] *= alpha; o[3][j] *= alpha;
            pshm[wave][j + 8 * half][l]      = f2bf(p0);
            pshm[wave][j + 8 * half][16 + l] = f2bf(p1);
        }
        // C-layout -> A-layout via per-wave LDS (same-wave DS ops are in-order)
        FragB pa;
        const unsigned short* pp = &pshm[wave][l][half * 8];
        pa.q[0] = *(const uint4*)(pp);
        pa.q[1] = *(const uint4*)(pp + 16);
#pragma unroll
        for (int t = 0; t < 4; ++t) {
            FragB vf;
            const unsigned short* vp = vb + (size_t)(t * 16 + l) * SEQ + kblk + half * 8;
            vf.q[0] = *(const uint4*)(vp);
            vf.q[1] = *(const uint4*)(vp + 16);
            o[t] = wmma_bf16(pa, vf, o[t]);
        }
    }

#pragma unroll
    for (int j = 0; j < 8; ++j) {
        float inv = 1.0f / lrun[j];
        int srow = qrow + j + 8 * half;
        size_t base = ((size_t)(b * SEQ + srow)) * DM + h * HD + l;
        aout[base +  0] = f2bf(o[0][j] * inv);
        aout[base + 16] = f2bf(o[1][j] * inv);
        aout[base + 32] = f2bf(o[2][j] * inv);
        aout[base + 48] = f2bf(o[3][j] * inv);
    }
}

// ---------------- output projection ----------------
// block 256 = 8 waves; wave tile 32x64; grid (1024/64, 4096/256)

__global__ __launch_bounds__(256) void k_gemm_out(
    const unsigned short* __restrict__ abf, const unsigned short* __restrict__ wT,
    const float* __restrict__ bo, float* __restrict__ out) {

    const int wave = threadIdx.x >> 5;
    const int lane = threadIdx.x & 31;
    const int l    = lane & 15;
    const int half = lane >> 4;
    const int row_base = blockIdx.y * 256 + wave * 32;
    const int n_base   = blockIdx.x * 64;

    const v8f zero = {0.f, 0.f, 0.f, 0.f, 0.f, 0.f, 0.f, 0.f};
    v8f c[2][4];
#pragma unroll
    for (int rt = 0; rt < 2; ++rt)
#pragma unroll
        for (int t = 0; t < 4; ++t) c[rt][t] = zero;

    for (int k = 0; k < DM; k += 32) {
        FragB a[2];
#pragma unroll
        for (int rt = 0; rt < 2; ++rt) {
            const unsigned short* ap =
                abf + (size_t)(row_base + rt * 16 + l) * DM + k + half * 8;
            a[rt].q[0] = *(const uint4*)(ap);
            a[rt].q[1] = *(const uint4*)(ap + 16);
        }
#pragma unroll
        for (int t = 0; t < 4; ++t) {
            FragB bf;
            const unsigned short* bp =
                wT + (size_t)(n_base + t * 16 + l) * DM + k + half * 8;
            bf.q[0] = *(const uint4*)(bp);
            bf.q[1] = *(const uint4*)(bp + 16);
            c[0][t] = wmma_bf16(a[0], bf, c[0][t]);
            c[1][t] = wmma_bf16(a[1], bf, c[1][t]);
        }
    }

    float bc[4];
#pragma unroll
    for (int t = 0; t < 4; ++t) bc[t] = bo[n_base + t * 16 + l];
#pragma unroll
    for (int rt = 0; rt < 2; ++rt) {
#pragma unroll
        for (int j = 0; j < 8; ++j) {
            int m = row_base + rt * 16 + j + 8 * half;
#pragma unroll
            for (int t = 0; t < 4; ++t)
                out[(size_t)m * DM + n_base + t * 16 + l] = c[rt][t][j] + bc[t];
        }
    }
}

// ---------------- host launch ----------------

extern "C" void kernel_launch(void* const* d_in, const int* in_sizes, int n_in,
                              void* d_out, int out_size, void* d_ws, size_t ws_size,
                              hipStream_t stream) {
    (void)in_sizes; (void)out_size; (void)ws_size;
    if (n_in < 9) return;

    const float* x  = (const float*)d_in[0];
    const float* Wq = (const float*)d_in[1];
    const float* bq = (const float*)d_in[2];
    const float* Wk = (const float*)d_in[3];
    const float* bk = (const float*)d_in[4];
    const float* Wv = (const float*)d_in[5];
    const float* bv = (const float*)d_in[6];
    const float* Wo = (const float*)d_in[7];
    const float* bo = (const float*)d_in[8];
    float* out = (float*)d_out;

    char* ws = (char*)d_ws;
    unsigned short* xbf   = (unsigned short*)ws; ws += (size_t)NROWS * DM * 2;     // 8 MB
    unsigned short* wqkvT = (unsigned short*)ws; ws += (size_t)3 * DM * DM * 2;    // 6 MB
    unsigned short* woT   = (unsigned short*)ws; ws += (size_t)DM * DM * 2;        // 2 MB
    unsigned short* qbf   = (unsigned short*)ws; ws += (size_t)NROWS * DM * 2;     // 8 MB
    unsigned short* kbf   = (unsigned short*)ws; ws += (size_t)NROWS * DM * 2;     // 8 MB
    unsigned short* vT    = (unsigned short*)ws; ws += (size_t)NROWS * DM * 2;     // 8 MB
    unsigned short* aout  = (unsigned short*)ws; ws += (size_t)NROWS * DM * 2;     // 8 MB

    k_cvt<<<(NROWS * DM) / 256, 256, 0, stream>>>(x, xbf, NROWS * DM);
    k_packT_qkv<<<(3 * DM * DM) / 256, 256, 0, stream>>>(Wq, Wk, Wv, wqkvT);
    k_packT<<<(DM * DM) / 256, 256, 0, stream>>>(Wo, woT);
    k_gemm_qkv<<<dim3(48, 16), 256, 0, stream>>>(xbf, wqkvT, bq, bk, bv, qbf, kbf, vT);
    k_attn<<<dim3(SEQ / 64, BATCH * NH), 128, 0, stream>>>(qbf, kbf, vT, aout);
    k_gemm_out<<<dim3(16, 16), 256, 0, stream>>>(aout, woT, bo, out);
}
